// TopMultiAttention_26431228740061
// MI455X (gfx1250) — matrix operand order repaired and checked by reading, not compiled, not run on hardware
//
#include <hip/hip_runtime.h>

typedef _Float16 half_t;
typedef __attribute__((ext_vector_type(16))) _Float16 v16h;
typedef __attribute__((ext_vector_type(8)))  _Float16 v8h;
typedef __attribute__((ext_vector_type(8)))  float    v8f;
typedef __attribute__((ext_vector_type(4)))  unsigned int v4u;
typedef __attribute__((ext_vector_type(8)))  unsigned int v8u;

#define BATCH 2
#define SEQ   2048
#define DM    1024
#define NH    16
#define DH    64
#define MTOT  (BATCH*SEQ)   /* 4096 */

// ---------------------------------------------------------------------------
// Feature probes (compile-safe: fall back to manual paths when absent)
// ---------------------------------------------------------------------------
#if defined(__has_builtin)
#  if __has_builtin(__builtin_amdgcn_global_load_async_to_lds_b128)
#    define HAVE_ASYNC_LDS 1
#  endif
#endif
#ifndef HAVE_ASYNC_LDS
#  define HAVE_ASYNC_LDS 0
#endif

#if HAVE_ASYNC_LDS
// Builtin is typed (v4i AS1*, v4i AS3*, imm offset, imm cpol)
typedef int v4i_b __attribute__((vector_size(16)));
typedef __attribute__((address_space(1))) v4i_b v4i_as1;
typedef __attribute__((address_space(3))) v4i_b v4i_as3;

static __device__ __forceinline__ void async_copy16(const void* g, void* l) {
  __builtin_amdgcn_global_load_async_to_lds_b128(
      (v4i_as1*)(uintptr_t)g,
      (v4i_as3*)(unsigned)(uintptr_t)l,
      /*offset=*/0, /*cpol=*/0);
}
static __device__ __forceinline__ void async_wait0() {
#if __has_builtin(__builtin_amdgcn_s_wait_asynccnt)
  __builtin_amdgcn_s_wait_asynccnt(0);
#else
  asm volatile("s_wait_asynccnt 0x0" ::: "memory");
#endif
}
#endif

static __device__ __forceinline__ void tensor_wait0() {
#if defined(__has_builtin) && __has_builtin(__builtin_amdgcn_s_wait_tensorcnt)
  __builtin_amdgcn_s_wait_tensorcnt(0);
#else
  asm volatile("s_wait_tensorcnt 0x0" ::: "memory");
#endif
}

// ---------------------------------------------------------------------------
// TDM: 2D f16 tile load, global -> LDS, with LDS row padding done by the DMA.
// D# built per CDNA5 ISA Group0/Group1 layouts.
//   pad_interval code 4 -> pad after every 32 DWORDs (=64 halfs = tile row)
//   pad_amount  code 3 -> insert 4 DWORDs (=8 halfs) -> LDS stride 72 halfs
// ---------------------------------------------------------------------------
static __device__ __forceinline__ void tdm_load_tile_f16(
    unsigned lds_byte_addr, unsigned long long gaddr,
    unsigned tensor_d0, unsigned tensor_d1,
    unsigned tile_d0, unsigned tile_d1, unsigned stride_d0,
    unsigned pad_interval_code, unsigned pad_amount_code) {
  v4u g0;
  g0[0] = 1u;                                         // count=1, user mode
  g0[1] = lds_byte_addr;                              // lds_addr
  g0[2] = (unsigned)(gaddr & 0xFFFFFFFFull);          // global_addr lo
  g0[3] = (unsigned)((gaddr >> 32) & 0x1FFFFFFull)    // global_addr hi (57b)
          | (2u << 30);                               // type = 2 ("image")
  v8u g1;
  g1[0] = (1u << 16)                                  // data_size = 2 bytes
        | (1u << 20)                                  // pad_enable
        | (pad_interval_code << 22)
        | (pad_amount_code << 25);
  g1[1] = (tensor_d0 & 0xFFFFu) << 16;                // tensor_dim0[15:0]
  g1[2] = ((tensor_d0 >> 16) & 0xFFFFu)               // tensor_dim0[31:16]
        | ((tensor_d1 & 0xFFFFu) << 16);              // tensor_dim1[15:0]
  g1[3] = ((tensor_d1 >> 16) & 0xFFFFu)               // tensor_dim1[31:16]
        | ((tile_d0 & 0xFFFFu) << 16);                // tile_dim0
  g1[4] = (tile_d1 & 0xFFFFu);                        // tile_dim1 (tile_dim2=0)
  g1[5] = stride_d0;                                  // tensor_dim0_stride lo32
  g1[6] = 0u;                                         // stride hi / dim1_stride lo
  g1[7] = 0u;
  asm volatile("tensor_load_to_lds %0, %1" :: "s"(g0), "s"(g1) : "memory");
}

// ---------------------------------------------------------------------------
// WMMA helpers (CDNA5: wave32, 16x16x32 f16 -> f32 accumulate)
// ---------------------------------------------------------------------------
static __device__ __forceinline__ v8f wmma_f16(v16h a, v16h b, v8f c) {
  return __builtin_amdgcn_wmma_f32_16x16x32_f16(
      false, a, false, b, (short)0, c, false, false);
}

// A fragment (16x32 f16 MxK): lane -> row L&15; kb=(L>>4)*8; two 16B chunks.
static __device__ __forceinline__ v16h lds_afrag(const half_t* base, int ld, int k0) {
  const int lane = threadIdx.x & 31;
  const half_t* p = base + (lane & 15) * ld + k0 + (lane >> 4) * 8;
  v8h lo = *(const v8h*)(p);
  v8h hi = *(const v8h*)(p + 16);
  return __builtin_shufflevector(lo, hi, 0,1,2,3,4,5,6,7,8,9,10,11,12,13,14,15);
}

// B fragment (32x16 f16 KxN) from LDS Bt[N][K]: lane -> col L&15; kb=(L>>4)*8; two 16B chunks.
static __device__ __forceinline__ v16h lds_bfrag(const half_t* baseT, int ld, int n0, int k0) {
  const int lane = threadIdx.x & 31;
  const half_t* p = baseT + (n0 + (lane & 15)) * ld + k0 + (lane >> 4) * 8;
  v8h lo = *(const v8h*)(p);
  v8h hi = *(const v8h*)(p + 16);
  return __builtin_shufflevector(lo, hi, 0,1,2,3,4,5,6,7,8,9,10,11,12,13,14,15);
}

static __device__ __forceinline__ v8f vzero8() {
  v8f z = {0.f,0.f,0.f,0.f,0.f,0.f,0.f,0.f};
  return z;
}

// ---------------------------------------------------------------------------
// Kernel 1: Qkv projection. C = A(f32) @ W(f32) via f16 WMMA; out f16 written
// in head layout [B, H, S, DH]. Tile 128x128, 8 waves, K-step 32.
// ---------------------------------------------------------------------------
__global__ __launch_bounds__(256)
void proj_in_kernel(const float* __restrict__ A, const float* __restrict__ W,
                    half_t* __restrict__ out) {
  __shared__ half_t As[128 * 40];
  __shared__ half_t Bt[128 * 40];
  const int m0   = blockIdx.x * 128;
  const int n0   = blockIdx.y * 128;
  const int tid  = threadIdx.x;
  const int wave = tid >> 5;
  const int lane = tid & 31;
  const int wm   = (wave >> 2) * 64;
  const int wn   = (wave & 3) * 32;

  v8f acc[4][2];
#pragma unroll
  for (int i = 0; i < 4; ++i)
#pragma unroll
    for (int j = 0; j < 2; ++j) acc[i][j] = vzero8();

  for (int k0 = 0; k0 < DM; k0 += 32) {
    __syncthreads();
#pragma unroll
    for (int i = 0; i < 4; ++i) {     // A tile f32 -> f16
      int f = tid + i * 256;
      int row = f >> 3, col = (f & 7) * 4;
      float4 v = *(const float4*)(A + (size_t)(m0 + row) * DM + k0 + col);
      half_t* d = As + row * 40 + col;
      d[0] = (half_t)v.x; d[1] = (half_t)v.y; d[2] = (half_t)v.z; d[3] = (half_t)v.w;
    }
#pragma unroll
    for (int i = 0; i < 4; ++i) {     // W tile f32 -> f16, transposed
      int f = tid + i * 256;
      int k = f >> 5, cn = (f & 31) * 4;
      float4 v = *(const float4*)(W + (size_t)(k0 + k) * DM + n0 + cn);
      Bt[(cn + 0) * 40 + k] = (half_t)v.x;
      Bt[(cn + 1) * 40 + k] = (half_t)v.y;
      Bt[(cn + 2) * 40 + k] = (half_t)v.z;
      Bt[(cn + 3) * 40 + k] = (half_t)v.w;
    }
    __syncthreads();

    v16h bf0 = lds_bfrag(Bt, 40, wn, 0);
    v16h bf1 = lds_bfrag(Bt, 40, wn + 16, 0);
#pragma unroll
    for (int i = 0; i < 4; ++i) {
      v16h af = lds_afrag(As + (wm + i * 16) * 40, 40, 0);
      acc[i][0] = wmma_f16(af, bf0, acc[i][0]);
      acc[i][1] = wmma_f16(af, bf1, acc[i][1]);
    }
  }

  // Straight-line head-layout store: n-group of 16 cols lies inside one head.
  const int cn = lane & 15;
  const int ch = lane >> 4;
#pragma unroll
  for (int i = 0; i < 4; ++i)
#pragma unroll
    for (int j = 0; j < 2; ++j) {
      int nbase = n0 + wn + j * 16;
      int h = nbase >> 6, dbase = (nbase & 63) + cn;
      int mbase = m0 + wm + i * 16 + 8 * ch;
      int b = mbase >> 11;
      half_t* o = out + (((size_t)(b * NH + h)) * SEQ + (mbase & 2047)) * DH + dbase;
#pragma unroll
      for (int r = 0; r < 8; ++r)
        o[(size_t)r * DH] = (half_t)acc[i][j][r];
    }
}

// ---------------------------------------------------------------------------
// Kernel 2: flash attention with post-softmax sigmoid gate.
// K tile staged by the Tensor Data Mover; Q tile by async-to-LDS.
// ---------------------------------------------------------------------------
__global__ __launch_bounds__(256)
void gated_attention_kernel(const half_t* __restrict__ Qh,
                            const half_t* __restrict__ Kh,
                            const half_t* __restrict__ Vh,
                            const float* __restrict__ rate_p,
                            half_t* __restrict__ Oh) {
  __shared__ half_t Qs[128 * 72];
  __shared__ half_t Ks[64 * 72];
  __shared__ half_t Vt[64 * 72];
  __shared__ half_t Ps[8 * 16 * 72];

  const int qt   = blockIdx.x;
  const int h    = blockIdx.y;
  const int b    = blockIdx.z;
  const int tid  = threadIdx.x;
  const int wave = tid >> 5;
  const int lane = tid & 31;
  const float rate = *rate_p;

  const size_t headoff = ((size_t)(b * NH + h)) * SEQ * DH;
  const half_t* Qg = Qh + headoff + (size_t)qt * 128 * DH;
  const half_t* Kg = Kh + headoff;
  const half_t* Vg = Vh + headoff;

  // ---- Stage Q tile (128x64 halfs) ----
#if HAVE_ASYNC_LDS
#pragma unroll
  for (int i = 0; i < 4; ++i) {
    int f = tid + i * 256;
    int row = f >> 3, col = (f & 7) * 8;
    async_copy16(Qg + (size_t)row * DH + col, Qs + row * 72 + col);
  }
  async_wait0();
#else
#pragma unroll
  for (int i = 0; i < 4; ++i) {
    int f = tid + i * 256;
    int row = f >> 3, col = (f & 7) * 8;
    *(v8h*)(Qs + row * 72 + col) = *(const v8h*)(Qg + (size_t)row * DH + col);
  }
#endif
  __syncthreads();

  const half_t* Qw = Qs + wave * 16 * 72;
  half_t*       Pw = Ps + wave * 16 * 72;
  v16h qa0 = lds_afrag(Qw, 72, 0);
  v16h qa1 = lds_afrag(Qw, 72, 32);

  float mrow[8], lrow[8];
  v8f acc[4];
#pragma unroll
  for (int r = 0; r < 8; ++r) { mrow[r] = -3.0e38f; lrow[r] = 0.f; }
#pragma unroll
  for (int g = 0; g < 4; ++g) acc[g] = vzero8();

  const int cn = lane & 15;
  const int ch = lane >> 4;

  for (int kb = 0; kb < SEQ; kb += 64) {
    __syncthreads();
    // ---- K tile via TDM (one descriptor, wave 0 only; pad -> stride 72) ----
    if (wave == 0) {
      tdm_load_tile_f16((unsigned)(uintptr_t)Ks,
                        (unsigned long long)(uintptr_t)(Kg + (size_t)kb * DH),
                        /*tensor_d0=*/DH, /*tensor_d1=*/SEQ,
                        /*tile_d0=*/DH, /*tile_d1=*/64, /*stride_d0=*/DH,
                        /*pad_interval(32 dw)=*/4, /*pad_amount(4 dw)=*/3);
    }
    // ---- V tile, software-transposed into Vt[d][key] ----
#pragma unroll
    for (int i = 0; i < 2; ++i) {
      int f = tid + i * 256;
      int row = f >> 3, col = (f & 7) * 8;
      v8h v = *(const v8h*)(Vg + (size_t)(kb + row) * DH + col);
#pragma unroll
      for (int j = 0; j < 8; ++j) Vt[(col + j) * 72 + row] = v[j];
    }
    if (wave == 0) tensor_wait0();
    __syncthreads();

    // ---- S = Q K^T / sqrt(dh) ----
    v8f sc[4];
#pragma unroll
    for (int g = 0; g < 4; ++g) {
      v8f c = vzero8();
      c = wmma_f16(qa0, lds_bfrag(Ks, 72, g * 16, 0), c);
      c = wmma_f16(qa1, lds_bfrag(Ks, 72, g * 16, 32), c);
      sc[g] = c;
    }
    const float scale = 0.125f;

    // ---- Online softmax with sigmoid gate ----
#pragma unroll
    for (int r = 0; r < 8; ++r) {
      float mx = -3.0e38f;
#pragma unroll
      for (int g = 0; g < 4; ++g) {
        sc[g][r] *= scale;
        mx = fmaxf(mx, sc[g][r]);
      }
#pragma unroll
      for (int msk = 8; msk >= 1; msk >>= 1) mx = fmaxf(mx, __shfl_xor(mx, msk, 16));

      float mnew  = fmaxf(mrow[r], mx);
      float alpha = __expf(mrow[r] - mnew);
      mrow[r] = mnew;

      float psum = 0.f;
#pragma unroll
      for (int g = 0; g < 4; ++g) {
        float s   = sc[g][r];
        float p   = __expf(s - mnew);
        psum += p;                               // ungated denominator
        float sig = 1.0f / (1.0f + __expf(-s));
        float pg  = (sig > rate) ? p : 0.f;      // gated numerator
        Pw[(r + 8 * ch) * 72 + g * 16 + cn] = (half_t)pg;
      }
#pragma unroll
      for (int msk = 8; msk >= 1; msk >>= 1) psum += __shfl_xor(psum, msk, 16);
      lrow[r] = lrow[r] * alpha + psum;
#pragma unroll
      for (int g = 0; g < 4; ++g) acc[g][r] *= alpha;
    }

    // ---- O += Pg @ V ----
    v16h pa0 = lds_afrag(Pw, 72, 0);
    v16h pa1 = lds_afrag(Pw, 72, 32);
#pragma unroll
    for (int g = 0; g < 4; ++g) {
      acc[g] = wmma_f16(pa0, lds_bfrag(Vt, 72, g * 16, 0), acc[g]);
      acc[g] = wmma_f16(pa1, lds_bfrag(Vt, 72, g * 16, 32), acc[g]);
    }
  }

  // ---- Normalize and store as [B, S, H, DH] ----
#pragma unroll
  for (int g = 0; g < 4; ++g)
#pragma unroll
    for (int r = 0; r < 8; ++r) {
      int q = qt * 128 + wave * 16 + r + 8 * ch;
      int d = g * 16 + cn;
      float v = acc[g][r] / lrow[r];
      Oh[(((size_t)b * SEQ + q) * NH + h) * DH + d] = (half_t)v;
    }
}

// ---------------------------------------------------------------------------
// Kernel 3: out(f32) = Ah(f16) @ Wo(f32->f16). A tile staged async-to-LDS.
// ---------------------------------------------------------------------------
__global__ __launch_bounds__(256)
void proj_out_kernel(const half_t* __restrict__ Ah, const float* __restrict__ W,
                     float* __restrict__ out) {
  __shared__ half_t As[128 * 40];
  __shared__ half_t Bt[128 * 40];
  const int m0   = blockIdx.x * 128;
  const int n0   = blockIdx.y * 128;
  const int tid  = threadIdx.x;
  const int wave = tid >> 5;
  const int lane = tid & 31;
  const int wm   = (wave >> 2) * 64;
  const int wn   = (wave & 3) * 32;

  v8f acc[4][2];
#pragma unroll
  for (int i = 0; i < 4; ++i)
#pragma unroll
    for (int j = 0; j < 2; ++j) acc[i][j] = vzero8();

  for (int k0 = 0; k0 < DM; k0 += 32) {
    __syncthreads();
    // A tile f16: 512 16B chunks — async straight into LDS when available
#if HAVE_ASYNC_LDS
#pragma unroll
    for (int i = 0; i < 2; ++i) {
      int f = tid + i * 256;
      int row = f >> 2, col = (f & 3) * 8;
      async_copy16(Ah + (size_t)(m0 + row) * DM + k0 + col, As + row * 40 + col);
    }
#else
#pragma unroll
    for (int i = 0; i < 2; ++i) {
      int f = tid + i * 256;
      int row = f >> 2, col = (f & 3) * 8;
      *(v8h*)(As + row * 40 + col) = *(const v8h*)(Ah + (size_t)(m0 + row) * DM + k0 + col);
    }
#endif
    // Wo tile f32 -> f16, transposed (needs VGPR conversion path)
#pragma unroll
    for (int i = 0; i < 4; ++i) {
      int f = tid + i * 256;
      int k = f >> 5, cnn = (f & 31) * 4;
      float4 v = *(const float4*)(W + (size_t)(k0 + k) * DM + n0 + cnn);
      Bt[(cnn + 0) * 40 + k] = (half_t)v.x;
      Bt[(cnn + 1) * 40 + k] = (half_t)v.y;
      Bt[(cnn + 2) * 40 + k] = (half_t)v.z;
      Bt[(cnn + 3) * 40 + k] = (half_t)v.w;
    }
#if HAVE_ASYNC_LDS
    async_wait0();
#endif
    __syncthreads();

    v16h bf0 = lds_bfrag(Bt, 40, wn, 0);
    v16h bf1 = lds_bfrag(Bt, 40, wn + 16, 0);
#pragma unroll
    for (int i = 0; i < 4; ++i) {
      v16h af = lds_afrag(As + (wm + i * 16) * 40, 40, 0);
      acc[i][0] = wmma_f16(af, bf0, acc[i][0]);
      acc[i][1] = wmma_f16(af, bf1, acc[i][1]);
    }
  }

  const int cn = lane & 15;
  const int ch = lane >> 4;
#pragma unroll
  for (int i = 0; i < 4; ++i)
#pragma unroll
    for (int j = 0; j < 2; ++j) {
      float* o = out + (size_t)(m0 + wm + i * 16 + 8 * ch) * DM + n0 + wn + j * 16 + cn;
#pragma unroll
      for (int r = 0; r < 8; ++r)
        o[(size_t)r * DM] = acc[i][j][r];
    }
}

// ---------------------------------------------------------------------------
// Host launcher
// ---------------------------------------------------------------------------
extern "C" void kernel_launch(void* const* d_in, const int* in_sizes, int n_in,
                              void* d_out, int out_size, void* d_ws, size_t ws_size,
                              hipStream_t stream) {
  (void)in_sizes; (void)n_in; (void)out_size; (void)ws_size;
  const float* X    = (const float*)d_in[0];
  const float* Y    = (const float*)d_in[1];
  const float* Z    = (const float*)d_in[2];
  const float* rate = (const float*)d_in[3];
  const float* Wq   = (const float*)d_in[4];
  const float* Wk   = (const float*)d_in[5];
  const float* Wv   = (const float*)d_in[6];
  const float* Wo   = (const float*)d_in[7];

  const size_t n_elem = (size_t)BATCH * NH * SEQ * DH;
  half_t* Qw = (half_t*)d_ws;
  half_t* Kw = Qw + n_elem;
  half_t* Vw = Kw + n_elem;
  half_t* Aw = Vw + n_elem;

  dim3 blk(256);
  dim3 gGemm(MTOT / 128, DM / 128);
  dim3 gAttn(SEQ / 128, NH, BATCH);

  proj_in_kernel<<<gGemm, blk, 0, stream>>>(X, Wq, Qw);
  proj_in_kernel<<<gGemm, blk, 0, stream>>>(Y, Wk, Kw);
  proj_in_kernel<<<gGemm, blk, 0, stream>>>(Z, Wv, Vw);
  gated_attention_kernel<<<gAttn, blk, 0, stream>>>(Qw, Kw, Vw, rate, Aw);
  proj_out_kernel<<<gGemm, blk, 0, stream>>>(Aw, Wo, (float*)d_out);
}
